// SVD_55387898249764
// MI455X (gfx1250) — matrix operand order repaired and checked by reading, not compiled, not run on hardware
//
#include <hip/hip_runtime.h>
#include <hip/hip_bf16.h>

typedef __attribute__((ext_vector_type(16))) _Float16 v16h;
typedef __attribute__((ext_vector_type(8)))  _Float16 v8h;
typedef __attribute__((ext_vector_type(4)))  _Float16 v4h;
typedef __attribute__((ext_vector_type(8)))  float    v8f;
typedef __attribute__((ext_vector_type(4)))  float    v4f;

#define BATCH      2048
#define NBUS       50000
#define NF         64
#define KSTEP      128
#define OUTER      391        // ceil(50000/128)
#define SPLITS     8
#define STEPS_PER  49         // ceil(391/8)
#define KPAD       50048      // 391*128, zero-padded K extent of yT
#define LDS_STRIDE 136        // 128 + 8 halves pad (breaks 16-way bank conflict)

union H16 { v16h v; v8h h[2]; };

static __device__ inline v8f wmma_f16(v16h a, v16h b, v8f c) {
  // D = A(16x32 f16) * B(32x16 f16) + C(16x16 f32)
  return __builtin_amdgcn_wmma_f32_16x16x32_f16(
      /*neg_a=*/false, a, /*neg_b=*/false, b,
      /*c_mod=*/(short)0, c, /*reuse_a=*/false, /*reuse_b=*/false);
}

// ---------------------------------------------------------------------------
// Kernel 1: y[50000][64] f32 -> yT[64][50048] f16 (transposed, zero-padded K)
// ---------------------------------------------------------------------------
__global__ __launch_bounds__(256) void svd_transpose_y(const float* __restrict__ y,
                                                       _Float16* __restrict__ yT) {
  int idx = blockIdx.x * 256 + threadIdx.x;     // grid sized exactly 64*KPAD/256
  int n = idx / KPAD;
  int k = idx - n * KPAD;
  _Float16 v = (_Float16)0.0f;
  if (k < NBUS) v = (_Float16)y[(size_t)k * NF + n];
  yT[idx] = v;
}

// ---------------------------------------------------------------------------
// One outer K-step: stream 16x128 f32 slab of R -> f16 LDS -> 4 WMMA steps.
// CHECK=true only for the single K-tail step (k0=49920) taken by split 7.
// ---------------------------------------------------------------------------
template <bool CHECK>
static __device__ inline void outer_step(int k0,
                                         const float* __restrict__ rrow,
                                         _Float16* __restrict__ sbase,
                                         const _Float16* __restrict__ abase,
                                         const _Float16* __restrict__ ybase,
                                         int lcol, const v16h& onesB,
                                         v8f& c, v8f& csum) {
  v4f r[4];
#pragma unroll
  for (int j = 0; j < 4; ++j) {
    int kk = k0 + (lcol + j * 8) * 4;   // 50000 % 4 == 0 -> whole-vector check
    if (CHECK) {
      v4f z = {};
      r[j] = (kk < NBUS) ? __builtin_nontemporal_load((const v4f*)(rrow + kk)) : z;
    } else {
      r[j] = __builtin_nontemporal_load((const v4f*)(rrow + kk));
    }
  }

  __syncthreads();                      // previous iter's LDS reads done
#pragma unroll
  for (int j = 0; j < 4; ++j) {
    *(v4h*)(sbase + j * 32) = __builtin_convertvector(r[j], v4h);  // exact 0/1
  }
  __syncthreads();                      // tile visible to all waves

#pragma unroll
  for (int s = 0; s < 4; ++s) {
    const int koff = s * 32;
    H16 a, b;
    a.h[0] = *(const v8h*)(abase + koff);
    a.h[1] = *(const v8h*)(abase + koff + 16);
    const _Float16* bp = ybase + k0 + koff;
    b.h[0] = *(const v8h*)bp;
    b.h[1] = *(const v8h*)(bp + 16);
    c    = wmma_f16(a.v, b.v, c);       // X partial
    csum = wmma_f16(a.v, onesB, csum);  // row-sum partial (every column == sum)
  }
}

// ---------------------------------------------------------------------------
// Kernel 2: split-K WMMA GEMM  X += R * y  (+ row-sums via all-ones B)
// grid = (128 M-tiles, 8 K-splits), block = 128 (4 waves; wave w -> N-tile w)
// ---------------------------------------------------------------------------
__global__ __launch_bounds__(128) void svd_gemm(const float* __restrict__ R,
                                                const _Float16* __restrict__ yT,
                                                float* __restrict__ xpart,
                                                float* __restrict__ spart) {
  __shared__ _Float16 ldsA[16 * LDS_STRIDE];

  const int mtile = blockIdx.x;          // 0..127
  const int split = blockIdx.y;          // 0..7
  const int tid   = threadIdx.x;         // 0..127
  const int lane  = tid & 31;
  const int wave  = tid >> 5;            // N-tile index 0..3

  const int step0 = split * STEPS_PER;
  int step1 = step0 + STEPS_PER;
  if (step1 > OUTER) step1 = OUTER;
  // only the very last outer step (k0=49920) has a K tail; it lives in split 7
  const bool hasTail = (step1 == OUTER);
  const int  fullEnd = hasTail ? (OUTER - 1) : step1;

  v8f c    = {};
  v8f csum = {};
  v16h onesB;
#pragma unroll
  for (int i = 0; i < 16; ++i) onesB[i] = (_Float16)1.0f;

  // cooperative R loader mapping: 16 rows x 128 cols f32, 4 x float4 per lane
  const int lrow = tid >> 3;             // 0..15
  const int lcol = tid & 7;              // float4 group within row
  const float* rrow = R + (size_t)(mtile * 16 + lrow) * NBUS;
  _Float16* sbase = ldsA + lrow * LDS_STRIDE + lcol * 4;

  // B fragment base: lane -> column n, half-wave -> K {0..7,16..23} / {8..15,24..31}
  const int ncol = wave * 16 + (lane & 15);
  const _Float16* ybase = yT + (size_t)ncol * KPAD + (lane >> 4) * 8;

  // A fragment base in LDS per ISA 16-bit A layout:
  // lanes 0-15: M=lane, K {0..7, 16..23}; lanes 16-31: M=lane-16, K {8..15, 24..31}
  const _Float16* abase = ldsA + (lane & 15) * LDS_STRIDE + (lane >> 4) * 8;

  for (int st = step0; st < fullEnd; ++st) {
    outer_step<false>(st * KSTEP, rrow, sbase, abase, ybase, lcol, onesB, c, csum);
  }
  if (hasTail) {                         // block-uniform: barriers stay converged
    outer_step<true>((OUTER - 1) * KSTEP, rrow, sbase, abase, ybase, lcol, onesB, c, csum);
  }

  // C layout: element j of lane l -> M = j + (l>=16 ? 8 : 0), N = l%16
  const int n    = lane & 15;
  const int mrow = (lane >> 4) * 8;
  float* xp = xpart + ((size_t)split * BATCH + mtile * 16 + mrow) * NF + wave * 16 + n;
#pragma unroll
  for (int j = 0; j < 8; ++j) xp[(size_t)j * NF] = c[j];

  if (wave == 0 && n == 0) {             // lanes 0 & 16 cover rows 0..7 / 8..15
    float* sp = spart + (size_t)split * BATCH + mtile * 16 + mrow;
#pragma unroll
    for (int j = 0; j < 8; ++j) sp[j] = csum[j];
  }
}

// ---------------------------------------------------------------------------
// Kernel 3: reduce K-splits, normalize, gather pu/qi, dot over 64 factors
// one wave32 per batch element, 2 factors per lane
// ---------------------------------------------------------------------------
__global__ __launch_bounds__(32) void svd_finalize(const float* __restrict__ xpart,
                                                   const float* __restrict__ spart,
                                                   const int* __restrict__ user,
                                                   const int* __restrict__ item,
                                                   const float* __restrict__ mean,
                                                   const float* __restrict__ bu,
                                                   const float* __restrict__ bi,
                                                   const float* __restrict__ qi,
                                                   const float* __restrict__ pu,
                                                   float* __restrict__ out) {
  const int b = blockIdx.x;
  const int t = threadIdx.x;             // 0..31

  float xs0 = 0.f, xs1 = 0.f, rs = 0.f;
#pragma unroll
  for (int s = 0; s < SPLITS; ++s) {
    const float* xp = xpart + ((size_t)s * BATCH + b) * NF;
    xs0 += xp[t];
    xs1 += xp[t + 32];
    rs  += spart[(size_t)s * BATCH + b];
  }
  const float inv = (rs > 0.f) ? (1.f / sqrtf(rs)) : 0.f;  // nan_to_num guard
  const float x0 = xs0 * inv, x1 = xs1 * inv;

  const int u  = user[b];
  const int it = item[b];
  float v = (pu[(size_t)u * NF + t]       + x0) * qi[(size_t)it * NF + t]
          + (pu[(size_t)u * NF + t + 32]  + x1) * qi[(size_t)it * NF + t + 32];
#pragma unroll
  for (int off = 16; off > 0; off >>= 1) v += __shfl_xor(v, off, 32);

  if (t == 0) out[b] = v + bu[u] + bi[it] + mean[0];
}

// ---------------------------------------------------------------------------
extern "C" void kernel_launch(void* const* d_in, const int* in_sizes, int n_in,
                              void* d_out, int out_size, void* d_ws, size_t ws_size,
                              hipStream_t stream) {
  const int*   user = (const int*)d_in[0];
  const int*   item = (const int*)d_in[1];
  const float* R    = (const float*)d_in[2];
  const float* mean = (const float*)d_in[3];
  const float* bu   = (const float*)d_in[4];
  const float* bi   = (const float*)d_in[5];
  const float* qi   = (const float*)d_in[6];
  const float* pu   = (const float*)d_in[7];
  const float* y    = (const float*)d_in[8];

  // workspace layout: xpart (4 MB) | spart (64 KB) | yT (6.4 MB)
  float*    xpart = (float*)d_ws;
  float*    spart = xpart + (size_t)SPLITS * BATCH * NF;
  _Float16* yT    = (_Float16*)(spart + (size_t)SPLITS * BATCH);

  svd_transpose_y<<<(NF * KPAD) / 256, 256, 0, stream>>>(y, yT);
  svd_gemm<<<dim3(128, SPLITS), 128, 0, stream>>>(R, yT, xpart, spart);
  svd_finalize<<<BATCH, 32, 0, stream>>>(xpart, spart, user, item, mean,
                                         bu, bi, qi, pu, (float*)d_out);
}
